// FanBackProjection2D_50422916055307
// MI455X (gfx1250) — matrix operand order repaired and checked by reading, not compiled, not run on hardware
//
#include <hip/hip_runtime.h>
#include <hip/hip_fp16.h>
#include <stdint.h>

typedef _Float16 v16h  __attribute__((ext_vector_type(16)));
typedef float    v8f   __attribute__((ext_vector_type(8)));
typedef unsigned int u32x4 __attribute__((ext_vector_type(4)));
typedef float    f32x4 __attribute__((ext_vector_type(4)));
typedef _Float16 f16x4 __attribute__((ext_vector_type(4)));

#define NB      16
#define NPROJ   720
#define NDET    1024
#define VY      512
#define VX      512
#define F16_STRIDE 1032      // halves per padded f16 row (2064B: conflict-free ds_load_b128)
#define SCALE_W 4096.0f      // keeps 1/r^2 weights in f16 normal range

__device__ __forceinline__ unsigned lds_off(const void* p) {
  // low 32 bits of a generic pointer into __shared__ == LDS byte offset
  return (unsigned)(uintptr_t)p;
}

__device__ __forceinline__ void wait_async0() {
  asm volatile("s_wait_asynccnt 0x0" ::: "memory");
}

// Stage one view's sinogram slab (16 batches x 1024 dets, f32, 64KB) into LDS
// with async copies tracked by ASYNCcnt. Thread t copies 64 consecutive floats.
__device__ __forceinline__ void async_copy_view(const float* __restrict__ sino,
                                                int p, float* f32buf, int t) {
#pragma unroll
  for (int i = 0; i < 16; ++i) {
    int e = t * 64 + i * 4;                 // flat float index in [0, 16384)
    int b = e >> 10;
    int d = e & 1023;
    const float* g = sino + (size_t)b * (NPROJ * NDET) + (size_t)p * NDET + d;
    unsigned lds = lds_off(f32buf + e);
    unsigned long long ga = (unsigned long long)(uintptr_t)g;
    asm volatile("global_load_async_to_lds_b128 %0, %1, off"
                 :: "v"(lds), "v"(ga) : "memory");
  }
}

__device__ __forceinline__ unsigned short f2h_bits(float x) {
  _Float16 h = (_Float16)x;
  return __builtin_bit_cast(unsigned short, h);
}

__device__ __forceinline__ int imin(int a, int b) { return a < b ? a : b; }
__device__ __forceinline__ int imax(int a, int b) { return a > b ? a : b; }

extern "C" __global__ __launch_bounds__(256)
void FanBackProjection2D_wmma(const float* __restrict__ sino,
                              const float* __restrict__ vol_origin,
                              const float* __restrict__ det_origin,
                              const float* __restrict__ vol_spacing,
                              const float* __restrict__ det_spacing,
                              const float* __restrict__ sidp,
                              const float* __restrict__ sddp,
                              const float* __restrict__ traj,
                              float* __restrict__ out) {
  extern __shared__ char smem[];
  float*     f32buf  = (float*)smem;                         // 65536 B
  _Float16*  f16buf0 = (_Float16*)(smem + 65536);            // 33024 B
  _Float16*  f16buf1 = f16buf0 + NB * F16_STRIDE;            // 33024 B

  const int t    = threadIdx.x;
  const int lane = t & 31;
  const int wv   = t >> 5;                // wave id 0..7
  const int x0   = blockIdx.x * 32;
  const int y0   = blockIdx.y * 32;

  const float oy   = vol_origin[0], ox = vol_origin[1];
  const float sy   = vol_spacing[0], sx = vol_spacing[1];
  const float dorg = det_origin[0];
  const float inv_dsp = 1.0f / det_spacing[0];
  const float sid  = sidp[0], sdd = sddp[0];

  const int n     = lane & 15;            // voxel-in-segment (B col) == batch (A row)
  const int hi16  = (lane >> 4) & 1;      // upper half-wave
  const int gbase = hi16 << 3;            // lane's first K-pair index in W operand

  v8f acc[8] = {};                        // 8 segments x 16x16 f32 accumulators

  async_copy_view(sino, 0, f32buf, t);    // prefetch view 0

  for (int p = 0; p < NPROJ; ++p) {
    float rx = traj[2 * p + 0];
    float ry = traj[2 * p + 1];

    wait_async0();                        // this wave's slab writes landed
    __syncthreads();                      // everyone's writes landed

    // convert f32 slab -> padded f16 staging buffer (double-buffered)
    _Float16* f16b = (p & 1) ? f16buf1 : f16buf0;
    {
      const f32x4* src = (const f32x4*)f32buf;
      int b  = t >> 4;
      int d0 = (t & 15) * 64;
      f16x4* dst = (f16x4*)(f16b + b * F16_STRIDE + d0);
#pragma unroll
      for (int i = 0; i < 16; ++i) {
        f32x4 v = src[t * 16 + i];
        dst[i] = __builtin_convertvector(v, f16x4);
      }
    }
    __syncthreads();                      // f16b ready; f32buf free to overwrite

    if (p + 1 < NPROJ) async_copy_view(sino, p + 1, f32buf, t);

    // loop-invariant A-row pointer (variable part per segment is just 2*base)
    const char* rowinv = (const char*)(f16b + n * F16_STRIDE) + (hi16 ? 32 : 0);

    // 8 voxel segments per wave: rows y0+4*wv .. +3, two 16-wide x halves each
#pragma unroll
    for (int s = 0; s < 8; ++s) {
      int row  = y0 + wv * 4 + (s >> 1);
      int half = s & 1;
      float px = (float)(x0 + half * 16 + n) * sx + ox;
      float py = (float)row * sy + oy;
      float dxv = px + sid * rx;          // voxel - source
      float dyv = py + sid * ry;
      float along = dxv * rx + dyv * ry;  // depth along central ray (>0)
      float sdet  = sdd * (dyv * rx - dxv * ry) * __builtin_amdgcn_rcpf(along);
      float r2    = dxv * dxv + dyv * dyv;
      float wd    = SCALE_W * __builtin_amdgcn_rcpf(r2);  // scaled distance weight
      float sidx  = (sdet - dorg) * inv_dsp;
      float fl    = floorf(sidx);
      int   i0    = (int)fl;
      float fr    = sidx - fl;
      float m0 = (i0 >= 0 && i0 < NDET) ? 1.0f : 0.0f;
      float m1 = (i0 + 1 >= 0 && i0 + 1 < NDET) ? 1.0f : 0.0f;
      float w0 = (1.0f - fr) * m0 * wd;
      float w1 = fr * m1 * wd;

      // window base: s is monotone in x along a row -> min at segment endpoints.
      // readlane keeps it scalar: base & the LDS address offset live in SGPRs.
      int iA = __builtin_amdgcn_readlane(i0, 0);
      int iB = __builtin_amdgcn_readlane(i0, 15);
      int base = imax(0, imin(imin(iA, iB), NDET - 32)) & ~7; // 16B-aligned window
      int k0 = i0 - base;                                     // tap halves k0, k0+1

      // Build sparse B operand (W: 32 det x 16 voxel, f16). Lane n holds
      // column n; dword j holds K-half pair g = gbase + j. Two taps land in
      // pair j0 (and j0+1 when k0 is odd): 2 cmp + 2 cndmask per dword.
      unsigned h0 = f2h_bits(w0);
      unsigned h1 = f2h_bits(w1);
      int j0  = k0 >> 1;
      int odd = k0 & 1;
      unsigned pack01 = h0 | (h1 << 16);
      unsigned Aval = odd ? (h0 << 16) : pack01;  // content of pair j0
      unsigned Bval = odd ? h1 : 0u;              // content of pair j0+1
      union { v16h v; unsigned u[8]; } Wb;
#pragma unroll
      for (int j = 0; j < 8; ++j) {
        int g = gbase + j;
        Wb.u[j] = (j0 == g) ? Aval : ((j0 + 1 == g) ? Bval : 0u);
      }

      // A operand (sino: 16 batch x 32 det window, f16) from LDS:
      // lane m holds batch row m, K blocks {0-15} (lo half) / {16-31}
      union { v16h v; u32x4 q[2]; } Am;
      const char* rowp = rowinv + 2 * base;
      Am.q[0] = *(const u32x4*)(rowp);
      Am.q[1] = *(const u32x4*)(rowp + 16);

      acc[s] = __builtin_amdgcn_wmma_f32_16x16x32_f16(
                   false, Am.v, false, Wb.v, (short)0, acc[s], false, false);
    }
  }

  // Epilogue: D layout -> out[b, y, x]; lane gives voxel col, VGPR r gives batch
  const float fs = sid * sdd * 3.14159265358979f / (float)NPROJ / SCALE_W;
#pragma unroll
  for (int s = 0; s < 8; ++s) {
    int row  = y0 + wv * 4 + (s >> 1);
    int x    = x0 + (s & 1) * 16 + n;
#pragma unroll
    for (int r = 0; r < 8; ++r) {
      int b = r + hi16 * 8;
      out[(size_t)b * (VY * VX) + (size_t)row * VX + x] = acc[s][r] * fs;
    }
  }
}

extern "C" void kernel_launch(void* const* d_in, const int* in_sizes, int n_in,
                              void* d_out, int out_size, void* d_ws, size_t ws_size,
                              hipStream_t stream) {
  (void)in_sizes; (void)n_in; (void)out_size; (void)d_ws; (void)ws_size;
  const float* sino        = (const float*)d_in[0];
  // d_in[1] = volume_shape (int64, static 512x512 — baked in)
  const float* vol_origin  = (const float*)d_in[2];
  const float* det_origin  = (const float*)d_in[3];
  const float* vol_spacing = (const float*)d_in[4];
  const float* det_spacing = (const float*)d_in[5];
  const float* sid         = (const float*)d_in[6];
  const float* sdd         = (const float*)d_in[7];
  const float* traj        = (const float*)d_in[8];
  float* out = (float*)d_out;

  dim3 grid(VX / 32, VY / 32);           // 16x16 blocks, 32x32 voxel tiles
  size_t smem = 65536 + 2 * (size_t)NB * F16_STRIDE * sizeof(_Float16);
  hipLaunchKernelGGL(FanBackProjection2D_wmma, grid, dim3(256), smem, stream,
                     sino, vol_origin, det_origin, vol_spacing, det_spacing,
                     sid, sdd, traj, out);
}
